// MSAColumnAttention_52183852646489
// MI455X (gfx1250) — matrix operand order repaired and checked by reading, not compiled, not run on hardware
//
#include <hip/hip_runtime.h>
#include <hip/hip_bf16.h>

typedef __attribute__((ext_vector_type(16))) _Float16     v16h;
typedef __attribute__((ext_vector_type(8)))  _Float16     v8h;
typedef __attribute__((ext_vector_type(8)))  float        v8f;
typedef __attribute__((ext_vector_type(8)))  unsigned int v8u;

#define WAVES_PER_BLOCK 8

__device__ __forceinline__ unsigned pack2h(float a, float b) {
  _Float16 ha = (_Float16)a, hb = (_Float16)b;
  unsigned short ua = __builtin_bit_cast(unsigned short, ha);
  unsigned short ub = __builtin_bit_cast(unsigned short, hb);
  return ((unsigned)ub << 16) | (unsigned)ua;
}

// Cheap sigmoid: prefer gfx1250 v_tanh_f32 (TRANS op), else v_rcp_f32.
// Avoids the IEEE div-scale/div-fixup sequence (~12 VALU ops/element).
__device__ __forceinline__ float fast_sigmoid(float x) {
#if __has_builtin(__builtin_amdgcn_tanhf)
  return __builtin_fmaf(0.5f, __builtin_amdgcn_tanhf(0.5f * x), 0.5f);
#else
  return __builtin_amdgcn_rcpf(1.0f + __expf(-x));
#endif
}

// Fused: LayerNorm -> (V = mn@Wv, G = sigmoid(mn@Wg)) -> (G*V)@Wo + bo
// (softmax over t sums to 1 against v[s], so the attention is an identity)
__global__ __launch_bounds__(256) void msa_fused_kernel(
    const float* __restrict__ m,     // (R, 32)
    const float* __restrict__ gamma, // (32,)
    const float* __restrict__ beta,  // (32,)
    const float* __restrict__ Wv,    // (32, 256)
    const float* __restrict__ Wg,    // (32, 256)
    const float* __restrict__ Wo,    // (256, 32)
    const float* __restrict__ bo,    // (32,)
    float* __restrict__ out,         // (R, 32)
    int numTiles)
{
  // B-operand layout in LDS: tile*512 + lane*16 + e, where lane = hi*16 + (col&15),
  // value = W[K = hi*16 + e][col]  (contiguous k order, same as the A operand)
  __shared__ _Float16 wv_lds[16 * 512];
  __shared__ _Float16 wg_lds[16 * 512];
  __shared__ _Float16 wo_lds[16 * 512];               // tile = kk*2 + no
  __shared__ _Float16 gvbuf[WAVES_PER_BLOCK * 512];   // per-wave 16x32 f16 strip

  const int tid = threadIdx.x;

  // ---- Stage Wv/Wg (32 x 256) into B layout, coalesced row reads ----
  for (int k = 0; k < 32; ++k) {
    float vv = Wv[k * 256 + tid];
    float gg = Wg[k * 256 + tid];
    int idx = ((tid >> 4) * 512) + (((((k >> 4) << 4) | (tid & 15))) * 16) + (k & 15);
    wv_lds[idx] = (_Float16)vv;
    wg_lds[idx] = (_Float16)gg;
  }
  // ---- Stage Wo (256 x 32): 8 rows x 32 cols per pass ----
  for (int p2 = 0; p2 < 32; ++p2) {
    int k = p2 * 8 + (tid >> 5);
    int n = tid & 31;
    float ov = Wo[k * 32 + n];
    int tI   = ((k >> 5) << 1) | (n >> 4);
    int kloc = k & 31;
    int idx  = tI * 512 + (((((kloc >> 4) << 4) | (n & 15))) * 16) + (kloc & 15);
    wo_lds[idx] = (_Float16)ov;
  }
  __syncthreads();

  const int lane   = tid & 31;
  const int waveId = tid >> 5;
  const int mrow   = lane & 15;   // row within tile (A) / col within tile (B,C,D)
  const int hi     = lane >> 4;
  _Float16* gbuf   = &gvbuf[waveId * 512];

  // gamma/beta for this lane's 16 channels: c = hi*16 + e
  float ga[16], be[16];
  {
    const float4* gp = (const float4*)(gamma + hi * 16);
    const float4* bp = (const float4*)(beta  + hi * 16);
    #pragma unroll
    for (int j = 0; j < 4; ++j) {
      float4 g4 = gp[j], b4 = bp[j];
      ga[4*j+0] = g4.x; ga[4*j+1] = g4.y; ga[4*j+2] = g4.z; ga[4*j+3] = g4.w;
      be[4*j+0] = b4.x; be[4*j+1] = b4.y; be[4*j+2] = b4.z; be[4*j+3] = b4.w;
    }
  }
  const float bo0 = bo[mrow];
  const float bo1 = bo[16 + mrow];

  const int waveGlobal = blockIdx.x * WAVES_PER_BLOCK + waveId;
  const int waveCount  = gridDim.x * WAVES_PER_BLOCK;

  for (int t = waveGlobal; t < numTiles; t += waveCount) {
    // ---- Load this lane's 16 contiguous channels of row (t*16 + mrow) ----
    const float* mp = m + ((size_t)(t * 16 + mrow)) * 32 + hi * 16;
    float x[16];
    #pragma unroll
    for (int j = 0; j < 4; ++j) {
      float4 x4 = ((const float4*)mp)[j];
      x[4*j+0] = x4.x; x[4*j+1] = x4.y; x[4*j+2] = x4.z; x[4*j+3] = x4.w;
    }
    // ---- LayerNorm over 32 channels (partner lane holds the other 16) ----
    float s = 0.f, q = 0.f;
    #pragma unroll
    for (int e = 0; e < 16; ++e) { s += x[e]; q += x[e] * x[e]; }
    s += __shfl_xor(s, 16, 32);
    q += __shfl_xor(q, 16, 32);
    float mean = s * 0.03125f;
    float var  = q * 0.03125f - mean * mean;
    float rs   = rsqrtf(var + 1e-5f);

    // normalize + f16 pack (channel order c = hi*16 + e)
    unsigned pk[8], qk[8];
    #pragma unroll
    for (int j = 0; j < 8; ++j) {
      float y0 = (x[2*j]   - mean) * rs * ga[2*j]   + be[2*j];
      float y1 = (x[2*j+1] - mean) * rs * ga[2*j+1] + be[2*j+1];
      pk[j] = pack2h(y0, y1);
    }
    #pragma unroll
    for (int j = 0; j < 8; ++j)
      qk[j] = (unsigned)__shfl_xor((int)pk[j], 16, 32);

    // Assemble A (16x32 f16): lanes hi=0 need K{0-15}, hi=1 need K{16-31}
    v8u au;
    #pragma unroll
    for (int j = 0; j < 4; ++j) {
      au[j]     = pk[j];
      au[4 + j] = pk[4 + j];
    }
    v16h a = __builtin_bit_cast(v16h, au);

    // ---- Fused GEMMs: per K-chunk of the output GEMM ----
    v8f acc0 = {}, acc1 = {};
    for (int kk = 0; kk < 8; ++kk) {
      #pragma unroll
      for (int nn = 0; nn < 2; ++nn) {
        int n = kk * 2 + nn;
        v16h bv = *(const v16h*)&wv_lds[n * 512 + lane * 16];
        v16h bg = *(const v16h*)&wg_lds[n * 512 + lane * 16];
        v8f cz = {};
        v8f vOut = __builtin_amdgcn_wmma_f32_16x16x32_f16(
            false, a, false, bv, (short)0, cz, false, false);
        v8f gOut = __builtin_amdgcn_wmma_f32_16x16x32_f16(
            false, a, false, bg, (short)0, cz, false, false);
        // gated value, spill strip to LDS (D-layout -> row-major 16x32)
        #pragma unroll
        for (int r = 0; r < 8; ++r) {
          float gv = vOut[r] * fast_sigmoid(gOut[r]);
          gbuf[(r + 8 * hi) * 32 + nn * 16 + mrow] = (_Float16)gv;
        }
      }
      // Reload strip as A operand (DS ops are in-order within a wave)
      v8h lo8 = *(const v8h*)&gbuf[mrow * 32 + hi * 16];
      v8h hi8 = *(const v8h*)&gbuf[mrow * 32 + hi * 16 + 8];
      v16h a2;
      #pragma unroll
      for (int e = 0; e < 8; ++e) { a2[e] = lo8[e]; a2[8 + e] = hi8[e]; }
      v16h b0 = *(const v16h*)&wo_lds[(kk * 2 + 0) * 512 + lane * 16];
      v16h b1 = *(const v16h*)&wo_lds[(kk * 2 + 1) * 512 + lane * 16];
      acc0 = __builtin_amdgcn_wmma_f32_16x16x32_f16(
          false, a2, false, b0, (short)0, acc0, false, false);
      acc1 = __builtin_amdgcn_wmma_f32_16x16x32_f16(
          false, a2, false, b1, (short)0, acc1, false, false);
    }

    // ---- Epilogue: + bo, store 16x32 f32 ----
    float* op = out + (size_t)(t * 16) * 32;
    #pragma unroll
    for (int r = 0; r < 8; ++r) {
      int row = r + 8 * hi;
      op[row * 32 + mrow]      = acc0[r] + bo0;
      op[row * 32 + 16 + mrow] = acc1[r] + bo1;
    }
  }
}

extern "C" void kernel_launch(void* const* d_in, const int* in_sizes, int n_in,
                              void* d_out, int out_size, void* d_ws, size_t ws_size,
                              hipStream_t stream) {
  (void)n_in; (void)out_size; (void)d_ws; (void)ws_size;
  const float* m     = (const float*)d_in[0];
  const float* gamma = (const float*)d_in[1];
  const float* beta  = (const float*)d_in[2];
  // d_in[3] = Wq, d_in[4] = Wk: mathematically dead (softmax over t sums to 1)
  const float* Wv = (const float*)d_in[5];
  const float* Wg = (const float*)d_in[6];
  const float* Wo = (const float*)d_in[7];
  const float* bo = (const float*)d_in[8];
  float* out = (float*)d_out;

  int rows     = in_sizes[0] / 32;          // B*s*i
  int numTiles = rows / 16;                 // 4096 for the reference shapes
  int grid = (numTiles + WAVES_PER_BLOCK - 1) / WAVES_PER_BLOCK;
  if (grid > 512) grid = 512;
  if (grid < 1) grid = 1;

  msa_fused_kernel<<<grid, 256, 0, stream>>>(m, gamma, beta, Wv, Wg, Wo, bo,
                                             out, numTiles);
}
